// MultiHeadNonLocalBlock2d_3100966387741
// MI455X (gfx1250) — matrix operand order repaired and checked by reading, not compiled, not run on hardware
//
#include <hip/hip_runtime.h>

#define BATCH  4
#define CCH    256
#define NPOS   4096
#define INTER_ 128
#define HEADS_ 4
#define DHEAD  32

typedef __attribute__((ext_vector_type(16))) _Float16 v16h;
typedef __attribute__((ext_vector_type(8)))  float    v8f;

union FragH {
  v16h v;
  unsigned int u[8];
  _Float16 h[16];
};

__device__ __forceinline__ void load16B(unsigned* dst, const void* src) {
  uint4 t = *(const uint4*)src;
  dst[0] = t.x; dst[1] = t.y; dst[2] = t.z; dst[3] = t.w;
}

// Direct memory->LDS async copy of one 16B chunk (gfx1250, ASYNCcnt-tracked).
// Low 32 bits of a generic pointer to __shared__ == LDS byte offset
// (flat aperture: ADDR[31:0] is the LDS address).
__device__ __forceinline__ void async_cp16(void* lds, const void* gaddr) {
  unsigned l = (unsigned)(uintptr_t)lds;
  asm volatile("global_load_async_to_lds_b128 %0, %1, off"
               :: "v"(l), "v"(gaddr) : "memory");
}

// ---------------------------------------------------------------------------
// Kernel 1: fused theta/phi/g 1x1-conv projections (f16 WMMA, K=256).
// Q,K written [b,h,n,d]; V written TRANSPOSED [b,h,d,n] so the attention
// kernel can stage it into LDS with straight async copies (no transpose).
// Scale 1/sqrt(D) folded into Q rows at write-out.
// ---------------------------------------------------------------------------
__global__ __launch_bounds__(256) void nlb_qkv_proj(
    const float* __restrict__ x,
    const float* __restrict__ w_theta, const float* __restrict__ b_theta,
    const float* __restrict__ w_phi,   const float* __restrict__ b_phi,
    const float* __restrict__ w_g,     const float* __restrict__ b_g,
    _Float16* __restrict__ qws, _Float16* __restrict__ kws,
    _Float16* __restrict__ vws)
{
  __shared__ _Float16 xT[64 * 40];   // [n][c], stride 40 halves (16B aligned)
  __shared__ _Float16 wl[32 * 40];   // [m][c]
  __shared__ float    bl[32];

  const int t    = threadIdx.x;
  const int wave = t >> 5;
  const int lane = t & 31;
  const int lmod = lane & 15;
  const int hi   = lane >> 4;
  const int mw   = wave >> 2;        // 0..1  (M sub-tile)
  const int nw   = wave & 3;         // 0..3  (N sub-tile)

  const int b  = blockIdx.z;
  const int m0 = blockIdx.y * 32;    // stacked row base, multiple of 32
  const int n0 = blockIdx.x * 64;

  const int which = m0 >> 7;         // 0=theta,1=phi,2=g
  const float* W  = (which == 0) ? w_theta : ((which == 1) ? w_phi : w_g);
  const float* Bi = (which == 0) ? b_theta : ((which == 1) ? b_phi : b_g);
  const int   mrel = m0 & 127;
  const int   head = mrel >> 5;
  const float osc  = (which == 0) ? 0.17677669529663687f : 1.0f;  // D^-0.5

  if (t < 32) bl[t] = Bi[mrel + t];

  v8f acc;
  #pragma unroll
  for (int r = 0; r < 8; ++r) acc[r] = 0.0f;

  for (int kt = 0; kt < 8; ++kt) {
    const int c0 = kt * 32;
    __syncthreads();
    // stage x^T tile [64 n][32 c] as f16 (global reads coalesced along n)
    #pragma unroll
    for (int j = 0; j < 8; ++j) {
      const int f  = t * 8 + j;
      const int cl = f >> 6;
      const int nl = f & 63;
      xT[nl * 40 + cl] =
          (_Float16)x[((long)b * CCH + (c0 + cl)) * NPOS + n0 + nl];
    }
    // stage weight tile [32 m][32 c] as f16
    if (t < 128) {
      #pragma unroll
      for (int j = 0; j < 8; ++j) {
        const int f  = t * 8 + j;
        const int ml = f >> 5;
        const int cl = f & 31;
        wl[ml * 40 + cl] = (_Float16)W[(mrel + ml) * CCH + c0 + cl];
      }
    }
    __syncthreads();

    FragH A, Bf;
    const int arow = mw * 16 + lmod;
    const int ak0  = hi ? 16 : 0;            // A layout: lanes>=16 hold K 16..31
    load16B(&A.u[0], &wl[arow * 40 + ak0]);
    load16B(&A.u[4], &wl[arow * 40 + ak0 + 8]);
    const int bn = nw * 16 + lmod;
    const int bk = hi ? 16 : 0;              // B layout: lanes>=16 hold K 16..31
    load16B(&Bf.u[0], &xT[bn * 40 + bk]);
    load16B(&Bf.u[4], &xT[bn * 40 + bk + 8]);
    acc = __builtin_amdgcn_wmma_f32_16x16x32_f16(false, A.v, false, Bf.v,
                                                 (short)0, acc, false, false);
  }

  const int n = n0 + nw * 16 + lmod;
  const long bh = (long)(b * HEADS_ + head);
  #pragma unroll
  for (int r = 0; r < 8; ++r) {
    const int d = mw * 16 + r + 8 * hi;      // C layout: VGPR r holds rows r / r+8
    const _Float16 val = (_Float16)((acc[r] + bl[d]) * osc);
    if (which == 2) {
      vws[(bh * DHEAD + d) * NPOS + n] = val;          // V: [b,h,d,n]
    } else if (which == 0) {
      qws[(bh * NPOS + n) * DHEAD + d] = val;          // Q: [b,h,n,d]
    } else {
      kws[(bh * NPOS + n) * DHEAD + d] = val;          // K: [b,h,n,d]
    }
  }
}

// ---------------------------------------------------------------------------
// Kernel 2: flash attention. 8 waves x 16 query rows / workgroup.
// 128-key tiles staged straight into LDS with GLOBAL_LOAD_ASYNC_TO_LDS_B128
// (V already transposed in memory); online softmax f32; 16 WMMAs per tile.
// ---------------------------------------------------------------------------
__global__ __launch_bounds__(256) void nlb_attn(
    const _Float16* __restrict__ qws, const _Float16* __restrict__ kws,
    const _Float16* __restrict__ vws, _Float16* __restrict__ yws)
{
  __shared__ _Float16 Kl[128 * 40];        // [key][d],  row stride 80B
  __shared__ _Float16 Vt[32 * 136];        // [d][key],  row stride 272B
  __shared__ _Float16 Pl[8][16 * 136];     // per-wave P scratch [qrow][key]

  const int t    = threadIdx.x;
  const int wave = t >> 5;
  const int lane = t & 31;
  const int lmod = lane & 15;
  const int hi   = lane >> 4;

  const int bh = blockIdx.y;
  const int n0 = blockIdx.x * 128 + wave * 16;

  const _Float16* Qb = qws + (long)bh * NPOS * DHEAD;
  const _Float16* Kb = kws + (long)bh * NPOS * DHEAD;
  const _Float16* Vb = vws + (long)bh * DHEAD * NPOS;   // [d][n]
  _Float16*       Yb = yws + (long)bh * NPOS * DHEAD;

  // Q A-fragment (already scaled by 1/sqrt(D) in kernel 1)
  FragH Aq;
  {
    const int qrow = n0 + lmod;
    const int ak0  = hi ? 16 : 0;
    load16B(&Aq.u[0], &Qb[(long)qrow * DHEAD + ak0]);
    load16B(&Aq.u[4], &Qb[(long)qrow * DHEAD + ak0 + 8]);
  }

  // async-copy index precompute
  const int krow  = t >> 1;            // 0..127   K tile row
  const int khalf = (t & 1) * 16;      // 0 / 16 halves within row
  const int vd0   = t >> 4;            // chunk t      -> d row 0..15
  const int vc0   = (t & 15) * 8;      // 16B chunk offset (halves)
  const int vd1   = (t + 256) >> 4;    // chunk t+256  -> d row 16..31
  const int vc1   = vc0;

  float mrun[8], lrun[8];
  v8f y0, y1;
  #pragma unroll
  for (int r = 0; r < 8; ++r) {
    mrun[r] = -1e30f; lrun[r] = 0.0f; y0[r] = 0.0f; y1[r] = 0.0f;
  }

  for (int tile = 0; tile < NPOS / 128; ++tile) {
    const int k0 = tile * 128;
    __syncthreads();   // all waves done reading previous tile's LDS

    // ---- direct memory->LDS async staging (4 x b128 per thread) ----
    async_cp16(&Kl[krow * 40 + khalf],     &Kb[(long)(k0 + krow) * DHEAD + khalf]);
    async_cp16(&Kl[krow * 40 + khalf + 8], &Kb[(long)(k0 + krow) * DHEAD + khalf + 8]);
    async_cp16(&Vt[vd0 * 136 + vc0],       &Vb[(long)vd0 * NPOS + k0 + vc0]);
    async_cp16(&Vt[vd1 * 136 + vc1],       &Vb[(long)vd1 * NPOS + k0 + vc1]);
    if (k0 + 128 < NPOS) {   // pull next tile toward L2 while this one computes
      __builtin_prefetch(&Kb[(long)(k0 + 128) * DHEAD], 0, 1);
      __builtin_prefetch(&Vb[(long)(k0 + 128)], 0, 1);
    }
    asm volatile("s_wait_asynccnt 0x0" ::: "memory");
    __syncthreads();   // everyone's async chunks visible

    // S = Q K^T for 8 key sub-tiles (one WMMA each: K-dim == D == 32)
    v8f s[8];
    #pragma unroll
    for (int j = 0; j < 8; ++j) {
      FragH Bk;
      const int col = j * 16 + lmod;
      const int kb  = hi ? 16 : 0;
      load16B(&Bk.u[0], &Kl[col * 40 + kb]);
      load16B(&Bk.u[4], &Kl[col * 40 + kb + 8]);
      v8f z;
      #pragma unroll
      for (int r = 0; r < 8; ++r) z[r] = 0.0f;
      s[j] = __builtin_amdgcn_wmma_f32_16x16x32_f16(false, Aq.v, false, Bk.v,
                                                    (short)0, z, false, false);
    }

    // online softmax: row r lives in VGPR r (lo half) / r+8 (hi half)
    #pragma unroll
    for (int r = 0; r < 8; ++r) {
      float tm = s[0][r];
      #pragma unroll
      for (int j = 1; j < 8; ++j) tm = fmaxf(tm, s[j][r]);
      #pragma unroll
      for (int off = 1; off < 16; off <<= 1)
        tm = fmaxf(tm, __shfl_xor(tm, off, 32));
      const float mn = fmaxf(mrun[r], tm);
      const float al = __expf(mrun[r] - mn);
      mrun[r] = mn;
      lrun[r] *= al;
      y0[r] *= al;
      y1[r] *= al;

      const int prow = r + 8 * hi;
      float ps = 0.0f;
      #pragma unroll
      for (int j = 0; j < 8; ++j) {
        const float p = __expf(s[j][r] - mn);
        ps += p;
        Pl[wave][prow * 136 + j * 16 + lmod] = (_Float16)p;
      }
      #pragma unroll
      for (int off = 1; off < 16; off <<= 1) ps += __shfl_xor(ps, off, 32);
      lrun[r] += ps;
    }

    // wave-local LDS transpose of P: make prior ds_stores visible to re-reads
    asm volatile("s_wait_dscnt 0x0" ::: "memory");

    // Y += P V  (4 K-steps of 32 keys, 2 N-tiles covering d=0..31)
    #pragma unroll
    for (int kq = 0; kq < 4; ++kq) {
      FragH Ap, Bv0, Bv1;
      const int pk = kq * 32 + (hi ? 16 : 0);
      load16B(&Ap.u[0], &Pl[wave][lmod * 136 + pk]);
      load16B(&Ap.u[4], &Pl[wave][lmod * 136 + pk + 8]);
      const int vk = kq * 32 + (hi ? 16 : 0);
      load16B(&Bv0.u[0], &Vt[lmod * 136 + vk]);
      load16B(&Bv0.u[4], &Vt[lmod * 136 + vk + 8]);
      load16B(&Bv1.u[0], &Vt[(16 + lmod) * 136 + vk]);
      load16B(&Bv1.u[4], &Vt[(16 + lmod) * 136 + vk + 8]);
      y0 = __builtin_amdgcn_wmma_f32_16x16x32_f16(false, Ap.v, false, Bv0.v,
                                                  (short)0, y0, false, false);
      y1 = __builtin_amdgcn_wmma_f32_16x16x32_f16(false, Ap.v, false, Bv1.v,
                                                  (short)0, y1, false, false);
    }
  }

  #pragma unroll
  for (int r = 0; r < 8; ++r) {
    const float inv = 1.0f / lrun[r];
    const long base = (long)(n0 + r + 8 * hi) * DHEAD;
    Yb[base + lmod]      = (_Float16)(y0[r] * inv);
    Yb[base + 16 + lmod] = (_Float16)(y1[r] * inv);
  }
}

// ---------------------------------------------------------------------------
// Kernel 3: z = w_z * y, then BN (b_z folded into shift) + residual.
// M = channel tile (rows), N = spatial (cols) -> coalesced fp32 stores.
// K loop over heads: i = h*32 + d blocks == WMMA K=32 steps.
// ---------------------------------------------------------------------------
__global__ __launch_bounds__(256) void nlb_out_proj(
    const _Float16* __restrict__ yws,
    const float* __restrict__ w_z, const float* __restrict__ b_z,
    const float* __restrict__ gamma, const float* __restrict__ beta,
    const float* __restrict__ mean,  const float* __restrict__ var,
    const float* __restrict__ x, float* __restrict__ out)
{
  __shared__ float sc[16], sh[16];

  const int t    = threadIdx.x;
  const int wave = t >> 5;
  const int lane = t & 31;
  const int lmod = lane & 15;
  const int hi   = lane >> 4;

  const int b  = blockIdx.z;
  const int c0 = blockIdx.y * 16;
  const int n0 = blockIdx.x * 128 + wave * 16;

  if (t < 16) {
    const float inv = gamma[c0 + t] * rsqrtf(var[c0 + t] + 1e-5f);
    sc[t] = inv;
    sh[t] = beta[c0 + t] + (b_z[c0 + t] - mean[c0 + t]) * inv;
  }
  __syncthreads();

  v8f acc;
  #pragma unroll
  for (int r = 0; r < 8; ++r) acc[r] = 0.0f;

  #pragma unroll
  for (int hh = 0; hh < 4; ++hh) {
    FragH A, Bf;
    const int arow = c0 + lmod;
    const int ak0  = hh * 32 + (hi ? 16 : 0);
    #pragma unroll
    for (int j = 0; j < 8; ++j)
      A.h[j] = (_Float16)w_z[arow * INTER_ + ak0 + j];
    #pragma unroll
    for (int j = 0; j < 8; ++j)
      A.h[8 + j] = (_Float16)w_z[arow * INTER_ + ak0 + 8 + j];

    const int n  = n0 + lmod;
    const int kb = hi ? 16 : 0;
    const _Float16* yp =
        yws + ((long)(b * HEADS_ + hh) * NPOS + n) * DHEAD + kb;
    load16B(&Bf.u[0], yp);
    load16B(&Bf.u[4], yp + 8);
    acc = __builtin_amdgcn_wmma_f32_16x16x32_f16(false, A.v, false, Bf.v,
                                                 (short)0, acc, false, false);
  }

  const int n = n0 + lmod;
  #pragma unroll
  for (int r = 0; r < 8; ++r) {
    const int cl = r + 8 * hi;
    const long idx = ((long)b * CCH + c0 + cl) * NPOS + n;
    out[idx] = acc[r] * sc[cl] + sh[cl] + x[idx];
  }
}

// ---------------------------------------------------------------------------
extern "C" void kernel_launch(void* const* d_in, const int* in_sizes, int n_in,
                              void* d_out, int out_size, void* d_ws,
                              size_t ws_size, hipStream_t stream) {
  (void)in_sizes; (void)n_in; (void)out_size; (void)ws_size;

  const float* x       = (const float*)d_in[0];
  const float* w_theta = (const float*)d_in[1];
  const float* b_theta = (const float*)d_in[2];
  const float* w_phi   = (const float*)d_in[3];
  const float* b_phi   = (const float*)d_in[4];
  const float* w_g     = (const float*)d_in[5];
  const float* b_g     = (const float*)d_in[6];
  const float* w_z     = (const float*)d_in[7];
  const float* b_z     = (const float*)d_in[8];
  const float* gamma   = (const float*)d_in[9];
  const float* beta    = (const float*)d_in[10];
  const float* mean    = (const float*)d_in[11];
  const float* var     = (const float*)d_in[12];
  float* out = (float*)d_out;

  const size_t seg =
      (size_t)BATCH * HEADS_ * NPOS * DHEAD * sizeof(_Float16);  // 4 MiB
  char* ws = (char*)d_ws;
  _Float16* qws = (_Float16*)(ws);
  _Float16* kws = (_Float16*)(ws + seg);
  _Float16* vws = (_Float16*)(ws + 2 * seg);
  _Float16* yws = (_Float16*)(ws + 3 * seg);

  nlb_qkv_proj<<<dim3(NPOS / 64, 384 / 32, BATCH), 256, 0, stream>>>(
      x, w_theta, b_theta, w_phi, b_phi, w_g, b_g, qws, kws, vws);
  nlb_attn<<<dim3(NPOS / 128, BATCH * HEADS_), 256, 0, stream>>>(
      qws, kws, vws, yws);
  nlb_out_proj<<<dim3(NPOS / 128, CCH / 16, BATCH), 256, 0, stream>>>(
      yws, w_z, b_z, gamma, beta, mean, var, x, out);
}
